// HAGAPS_83992380440678
// MI455X (gfx1250) — compile-verified, matrix operand order checked by reading, not run on hardware
//
#include <hip/hip_runtime.h>
#include <hip/hip_bf16.h>
#include <math.h>

typedef __attribute__((ext_vector_type(16))) __bf16 v16bf;
typedef __attribute__((ext_vector_type(8)))  float  v8f;

#define HDIM   128
#define NNODE  128128
#define NEDGE  512000
#define BGENE  128
#define L1     1001
#define LPAD1  1008
#define LC1    990      // conv1 out length
#define LP2IN  336      // padded conv2 input length (330 valid)
#define LC2    319      // conv2 out length
#define P2     79       // pooled length (L_CNN)
#define PKATT  96       // padded K for attention (79 -> 96)
#define FLATD  20224

// ---------------- WMMA fragment layout helpers (ISA 7.12.2) -----------------

// Swizzled-B element index for logical B[k][n] (K mult of 32, N mult of 16):
// fragment-contiguous so each lane loads its 16 bf16 elems as one 32B vector.
__device__ __host__ inline int swz_idx(int k, int n, int kchunks) {
  int nt = n >> 4, c = n & 15;
  int kc = k >> 5, kr = k & 31;
  int hf = (kr & 15) >> 3;
  int e  = (kr & 7) + ((kr & 16) ? 8 : 0);
  int lane = hf * 16 + c;
  return (((nt * kchunks + kc) * 32) + lane) * 16 + e;
}

__device__ inline v16bf load_frag_a(const __bf16* __restrict__ base, int ld) {
  // base -> A[row0][k0], row-major; lane holds row (lane&15), K half by lane>>4
  int lane = threadIdx.x & 31;
  int r = lane & 15, hf = lane >> 4;
  const __bf16* p = base + (long long)r * ld + hf * 8;
  v16bf f;
#pragma unroll
  for (int e = 0; e < 8; ++e) { f[e] = p[e]; f[8 + e] = p[16 + e]; }
  return f;
}

// ------- generic batched bf16 WMMA GEMM: C = A(MxK,row-major) * Bsw(KxN) ----
// Each wave computes a 16x64 strip: A fragment loaded once per K-chunk and
// reused against 4 B fragments (2.5 b128 loads per WMMA instead of 4).
// Launch with blockDim.x = (N/64)*32 (N must be a multiple of 64, <= 512).

__global__ void gemm_bf16_wmma(const __bf16* __restrict__ A,
                               const __bf16* __restrict__ Bsw,
                               float* __restrict__ C,
                               int M, int N, int K,
                               int lda, int ldc,
                               long long sA, long long sB, long long sC,
                               int Mvalid)
{
  int batch = blockIdx.z;
  const __bf16* Ab = A + (long long)batch * sA;
  const __bf16* Bb = Bsw + (long long)batch * sB;
  float* Cb = C + (long long)batch * sC;
  int mtile  = blockIdx.x;
  int wave   = threadIdx.x >> 5;
  int ntile0 = wave * 4;
  if (ntile0 * 16 >= N) return;
  int lane = threadIdx.x & 31;
  int r = lane & 15, hf = lane >> 4;
  int kchunks = K >> 5;
  long long tstride = (long long)kchunks * 512;  // elems per swizzled N-tile

  v8f acc0 = {0.f,0.f,0.f,0.f,0.f,0.f,0.f,0.f};
  v8f acc1 = acc0, acc2 = acc0, acc3 = acc0;
  const __bf16* ap = Ab + (long long)(mtile * 16) * lda;
  const __bf16* bp = Bb + (((long long)ntile0 * kchunks) * 32 + lane) * 16;
  for (int k0 = 0; k0 < K; k0 += 32) {
    v16bf a  = load_frag_a(ap + k0, lda);
    v16bf b0 = *(const v16bf*)(bp);
    v16bf b1 = *(const v16bf*)(bp + tstride);
    v16bf b2 = *(const v16bf*)(bp + 2 * tstride);
    v16bf b3 = *(const v16bf*)(bp + 3 * tstride);
    acc0 = __builtin_amdgcn_wmma_f32_16x16x32_bf16(false, a, false, b0, (short)0, acc0, false, false);
    acc1 = __builtin_amdgcn_wmma_f32_16x16x32_bf16(false, a, false, b1, (short)0, acc1, false, false);
    acc2 = __builtin_amdgcn_wmma_f32_16x16x32_bf16(false, a, false, b2, (short)0, acc2, false, false);
    acc3 = __builtin_amdgcn_wmma_f32_16x16x32_bf16(false, a, false, b3, (short)0, acc3, false, false);
    bp += 512;
  }
  v8f accs[4] = {acc0, acc1, acc2, acc3};
#pragma unroll
  for (int j = 0; j < 4; ++j) {
    int col = (ntile0 + j) * 16 + r;
#pragma unroll
    for (int q = 0; q < 8; ++q) {
      int m = mtile * 16 + hf * 8 + q;
      if (m < Mvalid) Cb[(long long)m * ldc + col] = accs[j][q];
    }
  }
}

// ---------------- implicit-GEMM 1D conv (kernel=12), WMMA -------------------
// act: (BGENE, Lpad, 128) bf16; wTsw: swizzled (K=1536, N=128); out f32.
// blockDim 64 (2 waves); each wave: 16 positions x 64 out channels.

__global__ void conv12_wmma(const __bf16* __restrict__ act,
                            const __bf16* __restrict__ wTsw,
                            const float* __restrict__ bias,
                            float* __restrict__ out,
                            int Lpad, int Lout)
{
  int gene  = blockIdx.y;
  int mtile = blockIdx.x;           // 16 output positions
  int wave  = threadIdx.x >> 5;     // 2 waves x 4 tiles -> 128 out channels
  int lane  = threadIdx.x & 31;
  int r = lane & 15, hf = lane >> 4;
  int l0 = mtile * 16;
  const long long tstride = 48LL * 512;   // elems per swizzled N-tile (kchunks=48)

  v8f acc0 = {0.f,0.f,0.f,0.f,0.f,0.f,0.f,0.f};
  v8f acc1 = acc0, acc2 = acc0, acc3 = acc0;
  const __bf16* bp = wTsw + (((long long)(wave * 4) * 48) * 32 + lane) * 16;
  for (int k = 0; k < 12; ++k) {
    const __bf16* abase = act + ((long long)gene * Lpad + l0 + k) * 128;
#pragma unroll
    for (int c = 0; c < 4; ++c) {
      v16bf a  = load_frag_a(abase + c * 32, 128);
      v16bf b0 = *(const v16bf*)(bp);
      v16bf b1 = *(const v16bf*)(bp + tstride);
      v16bf b2 = *(const v16bf*)(bp + 2 * tstride);
      v16bf b3 = *(const v16bf*)(bp + 3 * tstride);
      acc0 = __builtin_amdgcn_wmma_f32_16x16x32_bf16(false, a, false, b0, (short)0, acc0, false, false);
      acc1 = __builtin_amdgcn_wmma_f32_16x16x32_bf16(false, a, false, b1, (short)0, acc1, false, false);
      acc2 = __builtin_amdgcn_wmma_f32_16x16x32_bf16(false, a, false, b2, (short)0, acc2, false, false);
      acc3 = __builtin_amdgcn_wmma_f32_16x16x32_bf16(false, a, false, b3, (short)0, acc3, false, false);
      bp += 512;
    }
  }
  v8f accs[4] = {acc0, acc1, acc2, acc3};
#pragma unroll
  for (int j = 0; j < 4; ++j) {
    int col = (wave * 4 + j) * 16 + r;
    float bv = bias[col];
#pragma unroll
    for (int q = 0; q < 8; ++q) {
      int l = l0 + hf * 8 + q;
      if (l < Lout) out[((long long)gene * Lout + l) * 128 + col] = accs[j][q] + bv;
    }
  }
}

// ---------------- B-operand swizzle / conversion kernels --------------------

__global__ void swzB_f32(const float* __restrict__ in, __bf16* __restrict__ out,
                         int K, int N) { // in row-major (K,N)
  long long i = (long long)blockIdx.x * blockDim.x + threadIdx.x;
  if (i >= (long long)K * N) return;
  int k = (int)(i / N), n = (int)(i % N);
  out[swz_idx(k, n, K >> 5)] = (__bf16)in[i];
}

__global__ void swzBT_f32(const float* __restrict__ in, __bf16* __restrict__ out,
                          int K, int N) { // in row-major (N,K); B[k][n]=in[n*K+k]
  long long i = (long long)blockIdx.x * blockDim.x + threadIdx.x;
  if (i >= (long long)K * N) return;
  int n = (int)(i / K), k = (int)(i % K);
  out[swz_idx(k, n, K >> 5)] = (__bf16)in[i];
}

__global__ void swzB_conv(const float* __restrict__ w, __bf16* __restrict__ out) {
  // w (O=128, I=128, Kc=12); B[(kc*128+ii)][o] = w[o*1536 + ii*12 + kc]; K=1536
  long long i = (long long)blockIdx.x * blockDim.x + threadIdx.x;
  if (i >= 128LL * 128 * 12) return;
  int o = (int)(i / 1536);
  int rem = (int)(i % 1536);
  int ii = rem / 12, kc = rem % 12;
  out[swz_idx(kc * 128 + ii, o, 48)] = (__bf16)w[i];
}

// ---------------- small utility kernels -------------------------------------

__global__ void fillz_u32(unsigned int* p, long long n) {
  long long i = (long long)blockIdx.x * blockDim.x + threadIdx.x;
  if (i < n) p[i] = 0u;
}

__global__ void embed_k(const int* __restrict__ x, const float* __restrict__ emb,
                        float* __restrict__ h, __bf16* __restrict__ hb) {
  long long i = (long long)blockIdx.x * blockDim.x + threadIdx.x;
  if (i >= (long long)NNODE * 128) return;
  int n = (int)(i >> 7), c = (int)(i & 127);
  float v = emb[x[n] * 128 + c];
  h[i] = v; hb[i] = (__bf16)v;
}

__global__ void deg_k(const int* __restrict__ col, const float* __restrict__ w,
                      float* __restrict__ deg) {
  int e = blockIdx.x * blockDim.x + threadIdx.x;
  if (e < NEDGE) atomicAdd(&deg[col[e]], w[e]);
}

__global__ void dinv_k(float* deg) {
  int n = blockIdx.x * blockDim.x + threadIdx.x;
  if (n < NNODE) { float d = deg[n]; deg[n] = d > 0.f ? rsqrtf(d) : 0.f; }
}

__global__ void norm_k(const int* __restrict__ row, const int* __restrict__ col,
                       const float* __restrict__ w, const float* __restrict__ dinv,
                       float* __restrict__ nrm) {
  int e = blockIdx.x * blockDim.x + threadIdx.x;
  if (e < NEDGE) nrm[e] = dinv[row[e]] * w[e] * dinv[col[e]];
}

__global__ void scatter_k(const int* __restrict__ row, const int* __restrict__ col,
                          const float* __restrict__ nrm, const float* __restrict__ xw,
                          float* __restrict__ agg) {
  long long i = (long long)blockIdx.x * blockDim.x + threadIdx.x;
  if (i >= (long long)NEDGE * 128) return;
  int e = (int)(i >> 7), c = (int)(i & 127);
  atomicAdd(&agg[(long long)col[e] * 128 + c], nrm[e] * xw[(long long)row[e] * 128 + c]);
}

__global__ void arma_fin_k(const float* __restrict__ agg, const float* __restrict__ bias,
                           float* __restrict__ h, __bf16* __restrict__ hb) {
  long long i = (long long)blockIdx.x * blockDim.x + threadIdx.x;
  if (i >= (long long)NNODE * 128) return;
  float v = agg[i] + bias[(int)(i & 127)];
  v = v > 0.f ? v : 0.f;
  h[i] = v; hb[i] = (__bf16)v;
}

__global__ void build_act1_k(const float* __restrict__ h, __bf16* __restrict__ act) {
  long long i = (long long)blockIdx.x * blockDim.x + threadIdx.x;
  if (i >= (long long)BGENE * LPAD1 * 128) return;
  int c = (int)(i & 127);
  long long t = i >> 7;
  int l = (int)(t % LPAD1);
  int g = (int)(t / LPAD1);
  float v = (l < L1) ? h[((long long)g * L1 + l) * 128 + c] : 0.f;
  act[i] = (__bf16)v;
}

__device__ inline float lrelu01(float x) { return x > 0.f ? x : 0.01f * x; }

__global__ void pool1_k(const float* __restrict__ conv, const float* __restrict__ gamma,
                        const float* __restrict__ beta, __bf16* __restrict__ act2) {
  long long i = (long long)blockIdx.x * blockDim.x + threadIdx.x;
  if (i >= (long long)BGENE * LP2IN * 128) return;
  int c = (int)(i & 127);
  long long t = i >> 7;
  int p = (int)(t % LP2IN);
  int g = (int)(t / LP2IN);
  float out = 0.f;
  if (p < 330) {
    float sc = gamma[c] * rsqrtf(1.f + 1e-5f);
    float be = beta[c];
    float m = -1e30f;
    for (int j = 0; j < 3; ++j) {
      float v = conv[((long long)g * LC1 + (3 * p + j)) * 128 + c];
      v = lrelu01(v * sc + be);
      m = fmaxf(m, v);
    }
    out = m;
  }
  act2[i] = (__bf16)out;
}

__global__ void pool2_k(const float* __restrict__ conv, const float* __restrict__ gamma,
                        const float* __restrict__ beta, float* __restrict__ hp,
                        __bf16* __restrict__ hp_pad, __bf16* __restrict__ hpT_pad) {
  long long i = (long long)blockIdx.x * blockDim.x + threadIdx.x;
  if (i >= (long long)BGENE * P2 * 128) return;
  int c = (int)(i & 127);
  long long t = i >> 7;
  int p = (int)(t % P2);
  int g = (int)(t / P2);
  float sc = gamma[c] * rsqrtf(1.f + 1e-5f);
  float be = beta[c];
  float m = -1e30f;
  for (int j = 0; j < 4; ++j) {
    float v = conv[((long long)g * LC2 + (4 * p + j)) * 128 + c];
    v = lrelu01(v * sc + be);
    m = fmaxf(m, v);
  }
  hp[i] = m;
  hp_pad[((long long)g * PKATT + p) * 128 + c] = (__bf16)m;
  hpT_pad[((long long)g * 128 + c) * PKATT + p] = (__bf16)m;
}

__global__ void colsum_k(const float* __restrict__ hp, float* __restrict__ cs) {
  int i = blockIdx.x * blockDim.x + threadIdx.x; // p*128+c
  if (i >= P2 * 128) return;
  float s = 0.f;
  for (int g = 0; g < BGENE; ++g) s += hp[(long long)g * P2 * 128 + i];
  cs[i] = s;
}

__global__ void xcat_k(const float* __restrict__ hp, const float* __restrict__ cs,
                       __bf16* __restrict__ xcat) {
  long long i = (long long)blockIdx.x * blockDim.x + threadIdx.x;
  if (i >= (long long)BGENE * P2 * 128) return;
  int c = (int)(i & 127);
  long long t = i >> 7;       // g*P2 + p
  int p = (int)(t % P2);
  float v = hp[i];
  xcat[t * 256 + c] = (__bf16)v;
  xcat[t * 256 + 128 + c] = (__bf16)(cs[p * 128 + c] - v);
}

__device__ inline float sigf(float x) { return 1.f / (1.f + __expf(-x)); }

__global__ void gene_lstm_k(const float* __restrict__ xp, const float* __restrict__ Whh,
                            const float* __restrict__ bih, const float* __restrict__ bhh,
                            float* __restrict__ hg, __bf16* __restrict__ hg_swz) {
  __shared__ float sh[128], sc[128], sg[512];
  int b = blockIdx.x, tid = threadIdx.x; // 256 threads
  if (tid < 128) { sh[tid] = 0.f; sc[tid] = 0.f; }
  __syncthreads();
  for (int t = 0; t < P2; ++t) {
    const float* xpt = xp + ((long long)(b * P2 + t)) * 512;
    for (int gi = tid; gi < 512; gi += 256) {
      float v = xpt[gi] + bih[gi] + bhh[gi];
      const float* wr = Whh + (long long)gi * 128;
      for (int kk = 0; kk < 128; ++kk) v += wr[kk] * sh[kk];
      sg[gi] = v;
    }
    __syncthreads();
    if (tid < 128) {
      float ig = sigf(sg[tid]);
      float fg = sigf(sg[128 + tid]);
      float gg = tanhf(sg[256 + tid]);
      float og = sigf(sg[384 + tid]);
      float c = fg * sc[tid] + ig * gg;
      float h = og * tanhf(c);
      sc[tid] = c; sh[tid] = h;
      float r = h > 0.f ? h : 0.f;
      hg[((long long)(b * P2 + t)) * 128 + tid] = r;
      // direct write in swizzled-B layout (K=PKATT rows, kchunks=3)
      hg_swz[(long long)b * (PKATT * 128) + swz_idx(t, tid, PKATT / 32)] = (__bf16)r;
    }
    __syncthreads();
  }
}

__global__ void attn_softmax_k(const float* __restrict__ a, __bf16* __restrict__ wswz) {
  int b = blockIdx.x, g = threadIdx.x; // 128 threads: column g
  const float* ab = a + (long long)b * 128 * 128;
  float mx = -1e30f;
  for (int h = 0; h < 128; ++h) mx = fmaxf(mx, ab[h * 128 + g]);
  float s = 0.f;
  for (int h = 0; h < 128; ++h) s += __expf(ab[h * 128 + g] - mx);
  float inv = 1.f / s;
  __bf16* wb = wswz + (long long)b * 128 * 128;
  for (int h = 0; h < 128; ++h)
    wb[swz_idx(h, g, 4)] = (__bf16)(__expf(ab[h * 128 + g] - mx) * inv);
}

__global__ void build_ha_k(const float* __restrict__ hg, const float* __restrict__ bo,
                           __bf16* __restrict__ ha) {
  long long i = (long long)blockIdx.x * blockDim.x + threadIdx.x;
  if (i >= (long long)BGENE * P2 * 256) return;
  int f = (int)(i & 255);
  long long t = i >> 8;      // g*P2 + p
  float v = (f < 128) ? hg[t * 128 + f] : bo[t * 128 + (f - 128)];
  ha[i] = (__bf16)v;
}

__global__ void top_lstm_k(const float* __restrict__ xpf, const float* __restrict__ xpb,
                           const float* __restrict__ WhhF, const float* __restrict__ WhhB,
                           const float* __restrict__ bihF, const float* __restrict__ bhhF,
                           const float* __restrict__ bihB, const float* __restrict__ bhhB,
                           float* __restrict__ hh) {
  int dir = blockIdx.x, tid = threadIdx.x; // 256 threads
  const float* xp  = dir ? xpb : xpf;
  const float* Whh = dir ? WhhB : WhhF;
  const float* bi  = dir ? bihB : bihF;
  const float* bh  = dir ? bhhB : bhhF;
  __shared__ float sh[128], sc[128], sg[512];
  if (tid < 128) { sh[tid] = 0.f; sc[tid] = 0.f; }
  __syncthreads();
  for (int s = 0; s < BGENE; ++s) {
    int t = dir ? (BGENE - 1 - s) : s;
    const float* xpt = xp + (long long)t * 512;
    for (int gi = tid; gi < 512; gi += 256) {
      float v = xpt[gi] + bi[gi] + bh[gi];
      const float* wr = Whh + (long long)gi * 128;
      for (int kk = 0; kk < 128; ++kk) v += wr[kk] * sh[kk];
      sg[gi] = v;
    }
    __syncthreads();
    if (tid < 128) {
      float ig = sigf(sg[tid]);
      float fg = sigf(sg[128 + tid]);
      float gg = tanhf(sg[256 + tid]);
      float og = sigf(sg[384 + tid]);
      float c = fg * sc[tid] + ig * gg;
      float h = og * tanhf(c);
      sc[tid] = c; sh[tid] = h;
      hh[(long long)t * 256 + dir * 128 + tid] = lrelu01(h);
    }
    __syncthreads();
  }
}

__global__ void ffn_softmax_k(const float* __restrict__ hh, const float* __restrict__ W1,
                              const float* __restrict__ b1, const float* __restrict__ W2,
                              const float* __restrict__ b2, float* __restrict__ out) {
  __shared__ float sz[128];
  int t = threadIdx.x; // 128 threads, one gene each
  float acc2 = b2[0];
  const float* xrow = hh + (long long)t * 256;
  for (int j = 0; j < 32; ++j) {
    float v = b1[j];
    const float* w = W1 + j * 256;
    for (int kk = 0; kk < 256; ++kk) v += w[kk] * xrow[kk];
    acc2 += lrelu01(v) * W2[j];
  }
  sz[t] = acc2;
  __syncthreads();
  float mx = -1e30f;
  for (int g = 0; g < 128; ++g) mx = fmaxf(mx, sz[g]);
  float s = 0.f;
  for (int g = 0; g < 128; ++g) s += __expf(sz[g] - mx);
  out[t] = __expf(sz[t] - mx) / s;
}

// ---------------------------- launch ----------------------------------------

static inline int cdiv_ll(long long a, int b) { return (int)((a + b - 1) / b); }

extern "C" void kernel_launch(void* const* d_in, const int* in_sizes, int n_in,
                              void* d_out, int out_size, void* d_ws, size_t ws_size,
                              hipStream_t stream) {
  const int*   x     = (const int*)  d_in[0];
  const int*   eidx  = (const int*)  d_in[1];
  const float* eattr = (const float*)d_in[2];
  const float* emb   = (const float*)d_in[3];
  const float* a1Wi  = (const float*)d_in[4];
  const float* a1Wr  = (const float*)d_in[5];
  const float* a1b   = (const float*)d_in[6];
  const float* a2Wi  = (const float*)d_in[7];
  const float* a2Wr  = (const float*)d_in[8];
  const float* a2b   = (const float*)d_in[9];
  const float* c1w   = (const float*)d_in[10];
  const float* c1b   = (const float*)d_in[11];
  const float* c2w   = (const float*)d_in[12];
  const float* c2b   = (const float*)d_in[13];
  const float* bn1g  = (const float*)d_in[14];
  const float* bn1b  = (const float*)d_in[15];
  const float* bn2g  = (const float*)d_in[16];
  const float* bn2b  = (const float*)d_in[17];
  const float* gWih  = (const float*)d_in[18];
  const float* gWhh  = (const float*)d_in[19];
  const float* gbih  = (const float*)d_in[20];
  const float* gbhh  = (const float*)d_in[21];
  const float* fWih  = (const float*)d_in[22];
  const float* fWhh  = (const float*)d_in[23];
  const float* fbih  = (const float*)d_in[24];
  const float* fbhh  = (const float*)d_in[25];
  const float* bWih  = (const float*)d_in[26];
  const float* bWhh  = (const float*)d_in[27];
  const float* bbih  = (const float*)d_in[28];
  const float* bbhh  = (const float*)d_in[29];
  const float* W1    = (const float*)d_in[30];
  const float* b1    = (const float*)d_in[31];
  const float* W2    = (const float*)d_in[32];
  const float* b2    = (const float*)d_in[33];
  float* outp = (float*)d_out;

  char* wp = (char*)d_ws;
  auto carve = [&](size_t bytes) -> char* {
    char* r = wp;
    wp += (bytes + 255) & ~(size_t)255;
    return r;
  };
  const long long NH = (long long)NNODE * 128;
  float*  h     = (float*)carve(sizeof(float) * NH);
  float*  xw    = (float*)carve(sizeof(float) * NH);
  float*  agg   = (float*)carve(sizeof(float) * NH);
  float*  deg   = (float*)carve(sizeof(float) * NNODE);
  float*  nrm   = (float*)carve(sizeof(float) * NEDGE);
  __bf16* hb    = (__bf16*)carve(sizeof(__bf16) * NH);
  __bf16* wT1   = (__bf16*)carve(sizeof(__bf16) * 128 * 128 * 12);  // swizzled
  __bf16* wT2   = (__bf16*)carve(sizeof(__bf16) * 128 * 128 * 12);  // swizzled
  __bf16* bWi1  = (__bf16*)carve(sizeof(__bf16) * 16384);           // swizzled
  __bf16* bWr1  = (__bf16*)carve(sizeof(__bf16) * 16384);
  __bf16* bWi2  = (__bf16*)carve(sizeof(__bf16) * 16384);
  __bf16* bWr2  = (__bf16*)carve(sizeof(__bf16) * 16384);
  __bf16* gWihT = (__bf16*)carve(sizeof(__bf16) * 256 * 512);       // swizzled
  __bf16* fWihT = (__bf16*)carve(sizeof(__bf16) * (long long)FLATD * 512);
  __bf16* bWihT = (__bf16*)carve(sizeof(__bf16) * (long long)FLATD * 512);
  float*  hp    = (float*)carve(sizeof(float) * BGENE * P2 * 128);
  __bf16* hpPad = (__bf16*)carve(sizeof(__bf16) * BGENE * PKATT * 128);  // row-major A
  __bf16* hpTPad= (__bf16*)carve(sizeof(__bf16) * BGENE * 128 * PKATT);  // row-major A
  __bf16* hgSwz = (__bf16*)carve(sizeof(__bf16) * BGENE * PKATT * 128);  // swizzled B
  __bf16* xcat  = (__bf16*)carve(sizeof(__bf16) * (long long)BGENE * P2 * 256);
  __bf16* wSwz  = (__bf16*)carve(sizeof(__bf16) * (long long)BGENE * 128 * 128);
  float*  csum  = (float*)carve(sizeof(float) * P2 * 128);
  // lifetime-based aliases inside the big ARMA buffers:
  float*  conv1out = xw;                         // (128,990,128) f32
  __bf16* act1     = (__bf16*)agg;               // (128,1008,128) bf16
  __bf16* act2     = (__bf16*)hb;                // (128,336,128) bf16
  float*  conv2out = agg;                        // (128,319,128) f32 (act1 dead)
  float*  xp    = xw;                            // (10112,512) f32 (conv1out dead)
  float*  attA  = xw + 5177344;                  // (128,128,128) f32
  float*  bOut  = attA + 2097152;                // (128,79,128) f32
  float*  xpf   = bOut + 1294336;                // (128,512)
  float*  xpb   = xpf + 65536;                   // (128,512)
  float*  hhbuf = xpb + 65536;                   // (128,256)
  float*  hg    = agg;                           // (128,79,128) f32 (conv2out dead)
  __bf16* haB   = (__bf16*)(agg + 4194304);      // (128,20224) bf16, disjoint from hg

  const int TB = 256;
  const int TB128 = 64;   // GEMM block for N=128 (2 waves x 4 tiles)
  const int TB512 = 256;  // GEMM block for N=512 (8 waves x 4 tiles)
  // ---- weight conversions into swizzled-B fragment layout ----
  swzB_f32<<<cdiv_ll(16384, TB), TB, 0, stream>>>(a1Wi, bWi1, 128, 128);
  swzB_f32<<<cdiv_ll(16384, TB), TB, 0, stream>>>(a1Wr, bWr1, 128, 128);
  swzB_f32<<<cdiv_ll(16384, TB), TB, 0, stream>>>(a2Wi, bWi2, 128, 128);
  swzB_f32<<<cdiv_ll(16384, TB), TB, 0, stream>>>(a2Wr, bWr2, 128, 128);
  swzB_conv<<<cdiv_ll(128LL * 128 * 12, TB), TB, 0, stream>>>(c1w, wT1);
  swzB_conv<<<cdiv_ll(128LL * 128 * 12, TB), TB, 0, stream>>>(c2w, wT2);
  swzBT_f32<<<cdiv_ll(512LL * 256, TB), TB, 0, stream>>>(gWih, gWihT, 256, 512);
  swzBT_f32<<<cdiv_ll(512LL * FLATD, TB), TB, 0, stream>>>(fWih, fWihT, FLATD, 512);
  swzBT_f32<<<cdiv_ll(512LL * FLATD, TB), TB, 0, stream>>>(bWih, bWihT, FLATD, 512);
  // ---- zero pads / degree ----
  fillz_u32<<<cdiv_ll(NNODE, TB), TB, 0, stream>>>((unsigned int*)deg, NNODE);
  long long padw = (long long)BGENE * PKATT * 128 / 2; // bf16 pairs as u32
  fillz_u32<<<cdiv_ll(padw, TB), TB, 0, stream>>>((unsigned int*)hpPad, padw);
  fillz_u32<<<cdiv_ll(padw, TB), TB, 0, stream>>>((unsigned int*)hpTPad, padw);
  fillz_u32<<<cdiv_ll(padw, TB), TB, 0, stream>>>((unsigned int*)hgSwz, padw);
  // ---- embedding + graph norm ----
  embed_k<<<cdiv_ll(NH, TB), TB, 0, stream>>>(x, emb, h, hb);
  deg_k<<<cdiv_ll(NEDGE, TB), TB, 0, stream>>>(eidx + NEDGE, eattr, deg);
  dinv_k<<<cdiv_ll(NNODE, TB), TB, 0, stream>>>(deg);
  norm_k<<<cdiv_ll(NEDGE, TB), TB, 0, stream>>>(eidx, eidx + NEDGE, eattr, deg, nrm);
  // ---- ARMA layers ----
  const __bf16* aWi[2] = {bWi1, bWi2};
  const __bf16* aWr[2] = {bWr1, bWr2};
  const float*  aB[2]  = {a1b, a2b};
  for (int layer = 0; layer < 2; ++layer) {
    dim3 gg(NNODE / 16, 1, 1);
    gemm_bf16_wmma<<<gg, TB128, 0, stream>>>(hb, aWi[layer], xw, NNODE, 128, 128,
                                             128, 128, 0, 0, 0, NNODE);
    gemm_bf16_wmma<<<gg, TB128, 0, stream>>>(hb, aWr[layer], agg, NNODE, 128, 128,
                                             128, 128, 0, 0, 0, NNODE);
    scatter_k<<<cdiv_ll((long long)NEDGE * 128, TB), TB, 0, stream>>>(
        eidx, eidx + NEDGE, nrm, xw, agg);
    arma_fin_k<<<cdiv_ll(NH, TB), TB, 0, stream>>>(agg, aB[layer], h, hb);
  }
  // ---- CNN tower ----
  build_act1_k<<<cdiv_ll((long long)BGENE * LPAD1 * 128, TB), TB, 0, stream>>>(h, act1);
  conv12_wmma<<<dim3(62, BGENE, 1), TB128, 0, stream>>>(act1, wT1, c1b, conv1out, LPAD1, LC1);
  pool1_k<<<cdiv_ll((long long)BGENE * LP2IN * 128, TB), TB, 0, stream>>>(
      conv1out, bn1g, bn1b, act2);
  conv12_wmma<<<dim3(20, BGENE, 1), TB128, 0, stream>>>(act2, wT2, c2b, conv2out, LP2IN, LC2);
  pool2_k<<<cdiv_ll((long long)BGENE * P2 * 128, TB), TB, 0, stream>>>(
      conv2out, bn2g, bn2b, hp, hpPad, hpTPad);
  // ---- leave-one-out concat + gene LSTM ----
  colsum_k<<<cdiv_ll(P2 * 128, TB), TB, 0, stream>>>(hp, csum);
  xcat_k<<<cdiv_ll((long long)BGENE * P2 * 128, TB), TB, 0, stream>>>(hp, csum, xcat);
  gemm_bf16_wmma<<<dim3(BGENE * P2 / 16, 1, 1), TB512, 0, stream>>>(
      xcat, gWihT, xp, BGENE * P2, 512, 256, 256, 512, 0, 0, 0, BGENE * P2);
  gene_lstm_k<<<BGENE, TB, 0, stream>>>(xp, gWhh, gbih, gbhh, hg, hgSwz);
  // ---- attention ----
  gemm_bf16_wmma<<<dim3(8, 1, BGENE), TB128, 0, stream>>>(
      hpTPad, hgSwz, attA, 128, 128, PKATT, PKATT, 128,
      (long long)128 * PKATT, (long long)PKATT * 128, 128LL * 128, 128);
  attn_softmax_k<<<BGENE, 128, 0, stream>>>(attA, wSwz);
  gemm_bf16_wmma<<<dim3(PKATT / 16, 1, BGENE), TB128, 0, stream>>>(
      hpPad, wSwz, bOut, PKATT, 128, 128, 128, 128,
      (long long)PKATT * 128, 128LL * 128, (long long)P2 * 128, P2);
  build_ha_k<<<cdiv_ll((long long)BGENE * P2 * 256, TB), TB, 0, stream>>>(hg, bOut, haB);
  // ---- top bi-LSTM ----
  gemm_bf16_wmma<<<dim3(8, 1, 1), TB512, 0, stream>>>(
      haB, fWihT, xpf, BGENE, 512, FLATD, FLATD, 512, 0, 0, 0, BGENE);
  gemm_bf16_wmma<<<dim3(8, 1, 1), TB512, 0, stream>>>(
      haB, bWihT, xpb, BGENE, 512, FLATD, FLATD, 512, 0, 0, 0, BGENE);
  top_lstm_k<<<2, TB, 0, stream>>>(xpf, xpb, fWhh, bWhh, fbih, fbhh, bbih, bbhh, hhbuf);
  // ---- FFN + softmax over genes ----
  ffn_softmax_k<<<1, 128, 0, stream>>>(hhbuf, W1, b1, W2, b2, outp);
  (void)in_sizes; (void)n_in; (void)out_size; (void)ws_size;
}
